// SE3TransformerWrapper_4801773437164
// MI455X (gfx1250) — matrix operand order repaired and checked by reading, not compiled, not run on hardware
//
#include <hip/hip_runtime.h>

// MI455X / gfx1250, wave32. One workgroup per batch element.
// Live computation (NATTN=1 makes the h_r/h_l updates dead for the outputs):
//   h_l  = relu(phi(gather(hs_lig, idx)))          [4 x 32]
//   dots = h_r . h_l   via v_wmma_f32_16x16x32_f16 [4 x 4000]  (kept in LDS)
//   A    = softmax(dots) over the 4000 rec nodes
//   Yrec = (sum_i e_i * xyz_i) / (sum_i e_i)
// Memory-bound: ~160 MB total traffic -> ~7 us floor at 23.3 TB/s.

typedef __attribute__((ext_vector_type(16))) _Float16 v16h;
typedef __attribute__((ext_vector_type(8)))  float    v8f;

#define B_BATCH 256
#define NR      4000
#define NL      48
#define KQ      4
#define DIM     32
#define NTHREADS 256
#define NWAVES   8
#define NTILES   250          // 4000 / 16
#define DOTS_STRIDE 4008      // 4000 padded (mod 64 banks = 40) to avoid conflicts

// dynamic LDS layout (bytes)
#define OFF_DOTS   0                         // float [4 * 4008]      = 64128 B
#define OFF_BH     64128                     // _Float16 [512]        =  1024 B (32B aligned)
#define OFF_REDBIG 65152                     // float [256 * 16]      = 16384 B
#define OFF_RED    81536                     // float [256]           =  1024 B
#define OFF_MAXK   82560                     // float [4]
#define OFF_INVS   82576                     // float [4]
#define OFF_TOT    82592                     // float [16]
#define SMEM_BYTES 82688

extern "C" __global__ __launch_bounds__(NTHREADS)
void se3_attn_fused(const float* __restrict__ hs_rec,
                    const float* __restrict__ hs_lig,
                    const float* __restrict__ xyz_rec,
                    const float* __restrict__ phi_w,
                    const float* __restrict__ phi_b,
                    const int*   __restrict__ label_idx,
                    float* __restrict__ yrec_out,   // [B,4,3]
                    float* __restrict__ a_out)      // [B,4,4000]
{
    extern __shared__ char smem[];
    float*    dots   = (float*)   (smem + OFF_DOTS);
    _Float16* bh     = (_Float16*)(smem + OFF_BH);
    float*    redbig = (float*)   (smem + OFF_REDBIG);
    float*    red    = (float*)   (smem + OFF_RED);
    float*    maxk   = (float*)   (smem + OFF_MAXK);
    float*    invs   = (float*)   (smem + OFF_INVS);
    float*    totals = (float*)   (smem + OFF_TOT);

    const int b    = blockIdx.x;
    const int t    = threadIdx.x;
    const int lane = t & 31;
    const int wave = t >> 5;

    // ---------------- Phase A: h_l = relu(phi(gather(hs_lig))) ----------------
    // Packed as f16 WMMA B-matrix (32x16) in LDS; columns N>=4 are zero.
    // B order used here (k = 16h + s, same as A): lanes 0-15 hold col N=lane, K(d)=0..15;
    //                            lanes 16-31 hold col N=lane-16, K(d)=16..31.
    bh[t]       = (_Float16)0.f;
    bh[t + 256] = (_Float16)0.f;
    __syncthreads();
    if (t < KQ * DIM) {
        const int k = t >> 5;        // ligand query 0..3
        const int e = t & 31;        // output feature 0..31
        const int idx = label_idx[b * KQ + k];
        const float* row = hs_lig + ((size_t)b * NL + idx) * DIM;
        float acc = phi_b[e];
        #pragma unroll
        for (int d2 = 0; d2 < DIM; ++d2) acc += row[d2] * phi_w[e * DIM + d2];
        acc = fmaxf(acc, 0.f);
        if (e < 16) bh[k * 16 + e]               = (_Float16)acc;
        else        bh[(k + 16) * 16 + (e - 16)] = (_Float16)acc;
    }
    __syncthreads();

    // ---------------- Phase B: dots via v_wmma_f32_16x16x32_f16 ----------------
    // A operand (16x32 f16, order used here, same as B): lane<16 -> row M=lane,  K = 0..15
    //                                  lane>=16 -> row M=lane-16, K = 16..31
    const v16h bmat = *(const v16h*)&bh[lane * 16];
    const float* hr = hs_rec + (size_t)b * NR * DIM;
    const int mrow = lane & 15;
    const int koff = (lane < 16) ? 0 : 16;

    for (int tile = wave; tile < NTILES; tile += NWAVES) {   // wave-uniform: EXEC all-ones
        const int i0 = tile * 16;
        const float4* p = (const float4*)(hr + (size_t)(i0 + mrow) * DIM + koff);
        const float4 f0 = p[0];
        const float4 f1 = p[1];
        const float4 f2 = p[2];   // +8 floats
        const float4 f3 = p[3];
        v16h a;
        a[0] = (_Float16)f0.x; a[1] = (_Float16)f0.y; a[2]  = (_Float16)f0.z; a[3]  = (_Float16)f0.w;
        a[4] = (_Float16)f1.x; a[5] = (_Float16)f1.y; a[6]  = (_Float16)f1.z; a[7]  = (_Float16)f1.w;
        a[8] = (_Float16)f2.x; a[9] = (_Float16)f2.y; a[10] = (_Float16)f2.z; a[11] = (_Float16)f2.w;
        a[12] = (_Float16)f3.x; a[13] = (_Float16)f3.y; a[14] = (_Float16)f3.z; a[15] = (_Float16)f3.w;

        v8f c = {};
        c = __builtin_amdgcn_wmma_f32_16x16x32_f16(
                /*neg_a=*/false, a, /*neg_b=*/false, bmat,
                /*c_mod=*/(short)0, c, /*reuse_a=*/false, /*reuse_b=*/false);

        // D layout: lane holds column N=lane&15; c[r] = row M = r + (lane<16 ? 0 : 8)
        const int n = lane & 15;
        if (n < KQ) {
            const int mbase = i0 + ((lane < 16) ? 0 : 8);
            #pragma unroll
            for (int r = 0; r < 8; ++r)
                dots[n * DOTS_STRIDE + mbase + r] = c[r];
        }
    }
    __syncthreads();

    // ---------------- Phase C: per-k max over 4000 ----------------
    {
        const int k = t >> 6, j = t & 63;
        float m = -3.4e38f;
        for (int i = j; i < NR; i += 64) m = fmaxf(m, dots[k * DOTS_STRIDE + i]);
        red[t] = m;
        __syncthreads();
        #pragma unroll
        for (int s = 32; s > 0; s >>= 1) {
            if (j < s) red[t] = fmaxf(red[t], red[t + s]);
            __syncthreads();
        }
        if (t < KQ) maxk[t] = red[t * 64];
        __syncthreads();
    }

    // ------- Phase D: e = exp(dots - max); Σe and Σe*xyz (fused Yrec) -------
    {
        float mk[KQ];
        #pragma unroll
        for (int k = 0; k < KQ; ++k) mk[k] = maxk[k];
        float vsum[KQ] = {0.f, 0.f, 0.f, 0.f};
        float axz[KQ][3] = {};
        const float* xyz = xyz_rec + (size_t)b * NR * 3;
        for (int i = t; i < NR; i += NTHREADS) {
            const float x = xyz[i * 3 + 0];
            const float y = xyz[i * 3 + 1];
            const float z = xyz[i * 3 + 2];
            #pragma unroll
            for (int k = 0; k < KQ; ++k) {
                const int di = k * DOTS_STRIDE + i;
                const float e = __expf(dots[di] - mk[k]);
                dots[di] = e;
                vsum[k] += e;
                axz[k][0] += e * x;
                axz[k][1] += e * y;
                axz[k][2] += e * z;
            }
        }
        #pragma unroll
        for (int k = 0; k < KQ; ++k) {
            redbig[t * 16 + k * 4 + 0] = axz[k][0];
            redbig[t * 16 + k * 4 + 1] = axz[k][1];
            redbig[t * 16 + k * 4 + 2] = axz[k][2];
            redbig[t * 16 + k * 4 + 3] = vsum[k];
        }
        __syncthreads();
        if (t < 16) {
            float s = 0.f;
            for (int j = 0; j < NTHREADS; ++j) s += redbig[j * 16 + t];
            totals[t] = s;
        }
        __syncthreads();
        if (t < KQ) invs[t] = 1.0f / totals[t * 4 + 3];
        if (t < KQ * 3) {
            const int k = t / 3, c = t % 3;
            yrec_out[((size_t)b * KQ + k) * 3 + c] = totals[k * 4 + c] / totals[k * 4 + 3];
        }
        __syncthreads();
    }

    // ---------------- Phase E: A = e / Σe ----------------
    {
        float* aout = a_out + (size_t)b * KQ * NR;
        for (int j = t; j < KQ * NR; j += NTHREADS) {
            const int k = j / NR;
            const int i = j - k * NR;
            aout[j] = dots[k * DOTS_STRIDE + i] * invs[k];
        }
    }
}

extern "C" void kernel_launch(void* const* d_in, const int* in_sizes, int n_in,
                              void* d_out, int out_size, void* d_ws, size_t ws_size,
                              hipStream_t stream) {
    (void)in_sizes; (void)n_in; (void)out_size; (void)d_ws; (void)ws_size;
    const float* hs_rec    = (const float*)d_in[0];
    const float* hs_lig    = (const float*)d_in[1];
    const float* xyz_rec   = (const float*)d_in[2];
    const float* phi_w     = (const float*)d_in[3];
    const float* phi_b     = (const float*)d_in[4];
    // d_in[5], d_in[6] = ascaler1/2: dead for the outputs when NATTN == 1
    const int*   label_idx = (const int*)d_in[7];

    float* yrec = (float*)d_out;                       // [256,4,3]
    float* aout = yrec + (size_t)B_BATCH * KQ * 3;     // [256,4,4000]

    se3_attn_fused<<<B_BATCH, NTHREADS, SMEM_BYTES, stream>>>(
        hs_rec, hs_lig, xyz_rec, phi_w, phi_b, label_idx, yrec, aout);
}
